// Model38_65962107732448
// MI455X (gfx1250) — matrix operand order repaired and checked by reading, not compiled, not run on hardware
//
#include <hip/hip_runtime.h>
#include <hip/hip_bf16.h>
#include <math.h>

typedef __attribute__((ext_vector_type(16))) _Float16 v16h;
typedef __attribute__((ext_vector_type(8)))  float    v8f;

#define NTILES_W 16
#define WS_HALFS (NTILES_W*32*16)   // 8192 halfs = 16 KB
#define NBIAS    (6*32)             // q,k,v,op,rnn,dec  (f32)

union F16x2 { _Float16 h[2]; int i; };
union V16HU { int i[8]; v16h v; };
union F32I  { float f; int i; };

__device__ __forceinline__ int pack2h(float a, float b) {
  F16x2 u; u.h[0] = (_Float16)a; u.h[1] = (_Float16)b; return u.i;
}
// SWAPX16 (group-of-32, xor=0x10, and=0x1f): lanes n <-> n+16
__device__ __forceinline__ int swz16i(int v) {
  return __builtin_amdgcn_ds_swizzle(v, 0x401F);
}
__device__ __forceinline__ float swz16f(float x) {
  F32I u; u.f = x; u.i = swz16i(u.i); return u.f;
}

// ---- branch-free fast math (no EXEC-mask range splits) ----
__device__ __forceinline__ float fast_exp2(float x) {
#if __has_builtin(__builtin_amdgcn_exp2f)
  return __builtin_amdgcn_exp2f(x);
#else
  return exp2f(x);
#endif
}
__device__ __forceinline__ float fast_rcp(float x) {
#if __has_builtin(__builtin_amdgcn_rcpf)
  return __builtin_amdgcn_rcpf(x);
#else
  return 1.0f / x;
#endif
}
__device__ __forceinline__ float fast_tanh(float x) {
#if __has_builtin(__builtin_amdgcn_tanhf)
  return __builtin_amdgcn_tanhf(x);
#else
  // tanh(x) = 1 - 2/(2^(2*log2(e)*x) + 1); saturates cleanly at +-inf
  float e = fast_exp2(x * 2.8853900817779268f);
  return 1.0f - 2.0f * fast_rcp(e + 1.0f);
#endif
}
__device__ __forceinline__ float gelu_f(float x) {
  float u = 0.7978845608028654f * x * __builtin_fmaf(0.044715f, x * x, 1.0f);
  return 0.5f * x * (1.0f + fast_tanh(u));
}

// D(C-layout f32, 2 M-tiles) -> B-layout v16h for the next WMMA
__device__ __forceinline__ v16h d_to_b(v8f d0, v8f d1, bool lo) {
  int own[4], snd[4];
#pragma unroll
  for (int i = 0; i < 4; i++) {
    float oa = lo ? d0[2*i]   : d1[2*i];
    float ob = lo ? d0[2*i+1] : d1[2*i+1];
    float sa = lo ? d1[2*i]   : d0[2*i];
    float sb = lo ? d1[2*i+1] : d0[2*i+1];
    own[i] = pack2h(oa, ob);
    snd[i] = pack2h(sa, sb);
  }
  V16HU r;
#pragma unroll
  for (int i = 0; i < 4; i++) {
    int rcv = swz16i(snd[i]);
    r.i[i]     = lo ? own[i] : rcv;
    r.i[4 + i] = lo ? rcv    : own[i];
  }
  return r.v;
}

__device__ __forceinline__ v8f wmma_f16(v16h a, v16h b, v8f c) {
  return __builtin_amdgcn_wmma_f32_16x16x32_f16(false, a, false, b, (short)0, c,
                                                false, false);
}

// Weight A-tile from LDS; pointer carries an opaque 0 so loads stay in-loop.
__device__ __forceinline__ v16h ldA(const _Float16* swt, int tile, int lane) {
  return *(const v16h*)(swt + ((tile * 32 + lane) << 4));
}
// Bias (C-layout accumulator) from LDS.
__device__ __forceinline__ v8f ld_bias(const float* sbt, int mat, int mt, bool hi) {
  const float* p = sbt + mat * 32 + mt * 16 + (hi ? 8 : 0);
  v8f c;
#pragma unroll
  for (int r = 0; r < 8; r++) c[r] = p[r];
  return c;
}

// ---------------------------------------------------------------------------
// Kernel 0: weights (f32) -> f16 WMMA A-register layout in d_ws.
// A packing used here (contiguous k, same order as the B operands): lane L holds M=L%16; half j <-> K=j+(L>=16?16:0)
// Tiles: 0,1 emb(+bias in col5) | 2,3 wq | 4,5 wk | 6,7 wv | 8,9 out_proj |
//        10,11 w_ih | 12,13 w_hh | 14 dec(5x32 pad) | 15 head(3x5 pad, +bias col5)
// ---------------------------------------------------------------------------
__global__ void prep_weights(const float* __restrict__ emb_w,
                             const float* __restrict__ emb_b,
                             const float* __restrict__ in_proj_w,
                             const float* __restrict__ out_proj_w,
                             const float* __restrict__ w_ih,
                             const float* __restrict__ w_hh,
                             const float* __restrict__ dec_w,
                             const float* __restrict__ out_w,
                             const float* __restrict__ out_b,
                             _Float16* __restrict__ wsh) {
  int t = blockIdx.x;
  int lane = threadIdx.x;
  const float* W; const float* bias = nullptr;
  int nrows = 32, stride = 32, mt = 0;
  switch (t) {
    case 0: case 1:  W = emb_w; bias = emb_b; stride = 5; mt = t & 1; break;
    case 2: case 3:  W = in_proj_w;                       mt = t & 1; break;
    case 4: case 5:  W = in_proj_w + 32*32;               mt = t & 1; break;
    case 6: case 7:  W = in_proj_w + 64*32;               mt = t & 1; break;
    case 8: case 9:  W = out_proj_w;                      mt = t & 1; break;
    case 10: case 11: W = w_ih;                           mt = t & 1; break;
    case 12: case 13: W = w_hh;                           mt = t & 1; break;
    case 14:          W = dec_w; nrows = 5;               break;
    case 15: default: W = out_w; bias = out_b; nrows = 3; stride = 5; break;
  }
  int ncols = stride;
  int m = mt * 16 + (lane & 15);
#pragma unroll
  for (int j = 0; j < 16; j++) {
    int k = j + ((lane >= 16) ? 16 : 0);
    float v = 0.0f;
    if (m < nrows) {
      if (k < ncols)                 v = W[m * stride + k];
      else if (bias && k == ncols)   v = bias[m];   // folded bias column
    }
    wsh[((t * 32 + lane) << 4) + j] = (_Float16)v;
  }
}

// ---------------------------------------------------------------------------
// Main kernel: one wave = 16 batches/iter; token s -> N-tile s.
// ---------------------------------------------------------------------------
__global__ __launch_bounds__(256)
void model_main(const float* __restrict__ x,
                const _Float16* __restrict__ wsh,
                const float* __restrict__ in_proj_b,
                const float* __restrict__ out_proj_b,
                const float* __restrict__ b_ih,
                const float* __restrict__ b_hh,
                const float* __restrict__ fc_dec_b,
                float* __restrict__ out,
                int Bn, int ntiles) {
  __shared__ __align__(32) _Float16 sw[WS_HALFS];
  __shared__ __align__(32) float    sb[NBIAS];

  const int tid = threadIdx.x;
  for (int i = tid; i < WS_HALFS / 8; i += blockDim.x)
    ((int4*)sw)[i] = ((const int4*)wsh)[i];
  if (tid < NBIAS) {
    int blk = tid >> 5, i = tid & 31;
    float v;
    if (blk < 3)       v = in_proj_b[blk * 32 + i];     // q, k, v
    else if (blk == 3) v = out_proj_b[i];
    else if (blk == 4) v = b_ih[i] + b_hh[i];
    else               v = (i < 5) ? fc_dec_b[i] : 0.0f; // dec (padded)
    sb[tid] = v;
  }
  __syncthreads();

  const int lane = tid & 31;
  const bool hi = lane >= 16;
  const bool lo = !hi;
  const int  col = lane & 15;
  const float LOG2E = 1.4426950408889634f;
  v8f zero = {};

  const int gw = blockIdx.x * (blockDim.x >> 5) + (tid >> 5);
  const int nw = gridDim.x * (blockDim.x >> 5);

  for (int t = gw; t < ntiles; t += nw) {
    // Opaque zero: keeps every LDS weight/bias load inside the loop
    // (transient regs) instead of 350+ LICM-hoisted resident VGPRs.
    int zoff;
    asm volatile("s_mov_b32 %0, 0" : "=s"(zoff));
    const _Float16* swt = sw + zoff;
    const float*    sbt = sb + zoff;

    const int gb = t * 16 + col;
    const bool valid = gb < Bn;

    int tn = t + nw;
    if (tn < ntiles) __builtin_prefetch(x + (size_t)tn * 160, 0, 3);

    // ---- Bx (x^T, K padded 5->32, k=5 column = 1.0 for folded bias) ----
    v16h Bx[2];
#pragma unroll
    for (int nt = 0; nt < 2; nt++) {
      V16HU u;
#pragma unroll
      for (int i = 0; i < 8; i++) u.i[i] = 0;
      if (lo && valid) {
        const float* xr = x + (size_t)(gb * 2 + nt) * 5;
        u.i[0] = pack2h(xr[0], xr[1]);
        u.i[1] = pack2h(xr[2], xr[3]);
        u.i[2] = pack2h(xr[4], 1.0f);   // bias-1 in K=5
      }
      Bx[nt] = u.v;
    }

    // ---- embedding (bias folded into A), gelu, repack ----
    v16h Be[2];
#pragma unroll
    for (int nt = 0; nt < 2; nt++) {
      v8f e0 = wmma_f16(ldA(swt, 0, lane), Bx[nt], zero);
      v8f e1 = wmma_f16(ldA(swt, 1, lane), Bx[nt], zero);
#pragma unroll
      for (int r = 0; r < 8; r++) { e0[r] = gelu_f(e0[r]); e1[r] = gelu_f(e1[r]); }
      Be[nt] = d_to_b(e0, e1, lo);
    }

    // ---- k (kept), then q streamed through the score dot ----
    v8f kk[2][2];
#pragma unroll
    for (int nt = 0; nt < 2; nt++) {
      kk[nt][0] = wmma_f16(ldA(swt, 4, lane), Be[nt], ld_bias(sbt, 1, 0, hi));
      kk[nt][1] = wmma_f16(ldA(swt, 5, lane), Be[nt], ld_bias(sbt, 1, 1, hi));
    }
    float att[2][2];
#pragma unroll
    for (int sq = 0; sq < 2; sq++) {
      v8f q0 = wmma_f16(ldA(swt, 2, lane), Be[sq], ld_bias(sbt, 0, 0, hi));
      v8f q1 = wmma_f16(ldA(swt, 3, lane), Be[sq], ld_bias(sbt, 0, 1, hi));
      float sc[2];
#pragma unroll
      for (int sk = 0; sk < 2; sk++) {
        float p = 0.f;
#pragma unroll
        for (int r = 0; r < 8; r++)
          p += q0[r] * kk[sk][0][r] + q1[r] * kk[sk][1][r];
        sc[sk] = (p + swz16f(p)) * 0.17677669529663687f;  // 1/sqrt(32)
      }
      float m  = fmaxf(sc[0], sc[1]);
      float e0 = fast_exp2((sc[0] - m) * LOG2E);
      float e1 = fast_exp2((sc[1] - m) * LOG2E);
      float inv = fast_rcp(e0 + e1);
      att[sq][0] = e0 * inv; att[sq][1] = e1 * inv;
    }

    // ---- v ----
    v8f vv[2][2];
#pragma unroll
    for (int nt = 0; nt < 2; nt++) {
      vv[nt][0] = wmma_f16(ldA(swt, 6, lane), Be[nt], ld_bias(sbt, 2, 0, hi));
      vv[nt][1] = wmma_f16(ldA(swt, 7, lane), Be[nt], ld_bias(sbt, 2, 1, hi));
    }

    // ---- attn combine -> out_proj -> w_ih, per token ----
    v8f uD[2][2];
#pragma unroll
    for (int nt = 0; nt < 2; nt++) {
      v8f o0, o1;
#pragma unroll
      for (int r = 0; r < 8; r++) {
        o0[r] = att[nt][0] * vv[0][0][r] + att[nt][1] * vv[1][0][r];
        o1[r] = att[nt][0] * vv[0][1][r] + att[nt][1] * vv[1][1][r];
      }
      v16h Bo = d_to_b(o0, o1, lo);
      v8f p0 = wmma_f16(ldA(swt, 8, lane), Bo, ld_bias(sbt, 3, 0, hi));
      v8f p1 = wmma_f16(ldA(swt, 9, lane), Bo, ld_bias(sbt, 3, 1, hi));
      v16h Bo2 = d_to_b(p0, p1, lo);
      uD[nt][0] = wmma_f16(ldA(swt, 10, lane), Bo2, ld_bias(sbt, 4, 0, hi));
      uD[nt][1] = wmma_f16(ldA(swt, 11, lane), Bo2, ld_bias(sbt, 4, 1, hi));
    }

    // ---- RNN: h1 = tanh(u0); h2 = tanh(u1 + h1 @ w_hh^T) ----
    v8f h1a, h1b;
#pragma unroll
    for (int r = 0; r < 8; r++) {
      h1a[r] = fast_tanh(uD[0][0][r]);
      h1b[r] = fast_tanh(uD[0][1][r]);
    }
    v16h Bh1 = d_to_b(h1a, h1b, lo);
    v8f g0 = wmma_f16(ldA(swt, 12, lane), Bh1, zero);
    v8f g1 = wmma_f16(ldA(swt, 13, lane), Bh1, zero);
    v8f h2a, h2b;
#pragma unroll
    for (int r = 0; r < 8; r++) {
      h2a[r] = fast_tanh(uD[1][0][r] + g0[r]);
      h2b[r] = fast_tanh(uD[1][1][r] + g1[r]);
    }

    // ---- decode + head ----
    v16h Bh2 = d_to_b(h2a, h2b, lo);
    v8f dv = wmma_f16(ldA(swt, 14, lane), Bh2, ld_bias(sbt, 5, 0, hi));
#pragma unroll
    for (int r = 0; r < 8; r++) dv[r] = gelu_f(dv[r]);
    dv[5] = 1.0f;                         // bias-1 for folded head bias (K=5)
    v16h Bd = d_to_b(dv, zero, lo);
    v8f ov = wmma_f16(ldA(swt, 15, lane), Bd, zero);

    if (lo && valid) {
      float* op = out + (size_t)gb * 3;
      op[0] = ov[0]; op[1] = ov[1]; op[2] = ov[2];
    }
  }
}

extern "C" void kernel_launch(void* const* d_in, const int* in_sizes, int n_in,
                              void* d_out, int out_size, void* d_ws, size_t ws_size,
                              hipStream_t stream) {
  const float* x          = (const float*)d_in[0];
  const float* fc_emb_w   = (const float*)d_in[1];
  const float* fc_emb_b   = (const float*)d_in[2];
  const float* in_proj_w  = (const float*)d_in[3];
  const float* in_proj_b  = (const float*)d_in[4];
  const float* out_proj_w = (const float*)d_in[5];
  const float* out_proj_b = (const float*)d_in[6];
  const float* w_ih       = (const float*)d_in[7];
  const float* b_ih       = (const float*)d_in[8];
  const float* w_hh       = (const float*)d_in[9];
  const float* b_hh       = (const float*)d_in[10];
  const float* fc_dec_w   = (const float*)d_in[11];
  const float* fc_dec_b   = (const float*)d_in[12];
  const float* out_w      = (const float*)d_in[13];
  const float* out_b      = (const float*)d_in[14];
  float* out = (float*)d_out;
  _Float16* wsh = (_Float16*)d_ws;

  int Bn = in_sizes[0] / 10;
  int ntiles = (Bn + 15) / 16;

  prep_weights<<<NTILES_W, 32, 0, stream>>>(fc_emb_w, fc_emb_b, in_proj_w,
                                            out_proj_w, w_ih, w_hh, fc_dec_w,
                                            out_w, out_b, wsh);

  int blocks = 2048;
  int need = (ntiles + 7) / 8;
  if (blocks > need) blocks = need;
  if (blocks < 1) blocks = 1;
  model_main<<<blocks, 256, 0, stream>>>(x, wsh, in_proj_b, out_proj_b,
                                         b_ih, b_hh, fc_dec_b, out, Bn, ntiles);
}
